// DTMJax_47760036331638
// MI455X (gfx1250) — matrix operand order repaired and checked by reading, not compiled, not run on hardware
//
#include <hip/hip_runtime.h>
#include <hip/hip_bf16.h>

// CDNA5 / gfx1250 vector types for WMMA
typedef __attribute__((ext_vector_type(16))) _Float16 v16h;
typedef __attribute__((ext_vector_type(8)))  float    v8f;

#define KTOPIC 64
#define INIT_ALPHA (50.0f / 64.0f)

// ---------------------------------------------------------------------------
// Kernel 1: zero the whole output block (111 MB) — pure streaming stores.
// ---------------------------------------------------------------------------
__global__ void dtm_zero_kernel(float* __restrict__ out, int n) {
    int n4 = n >> 2;
    int i = blockIdx.x * blockDim.x + threadIdx.x;
    int stride = gridDim.x * blockDim.x;
    float4* o4 = reinterpret_cast<float4*>(out);
    for (int j = i; j < n4; j += stride) {
        o4[j] = float4{0.0f, 0.0f, 0.0f, 0.0f};
    }
    // tail (out_size is 4-divisible in practice; keep robust anyway)
    for (int j = (n4 << 2) + i; j < n; j += stride) out[j] = 0.0f;
}

// ---------------------------------------------------------------------------
// Kernel 2: scatter-adds.
//  - CDK / CWK / eta: direct global_atomic_add_f32 (low contention, L2-resident)
//  - CK (512 slots, 4M adds): one-hot outer-product histogram on the matrix
//    unit: 4x v_wmma_f32_16x16x32_f16 per 32-word batch, wave-private f32
//    accumulators, then LDS (ds_add_f32) block reduce, then global flush.
// ---------------------------------------------------------------------------
__global__ void __launch_bounds__(256)
dtm_scatter_kernel(const int*   __restrict__ tArr,   // time_ind_per_word
                   const int*   __restrict__ dArr,   // doc_indexes
                   const int*   __restrict__ wArr,   // flatW
                   const float* __restrict__ nArr,   // N_per_word
                   const int*   __restrict__ zArr,   // flatZ
                   float* __restrict__ outCDK,
                   float* __restrict__ outCWK,
                   float* __restrict__ outCK,
                   float* __restrict__ outEta,
                   int AW, int cwkPlaneStride /* = V*K */) {
    __shared__ float sCK[8 * KTOPIC];   // [T=8][K=64] block-local CK tile

    const int tid        = threadIdx.x;
    const int lane       = tid & 31;
    const int waveInBlk  = tid >> 5;
    const int wavesPerBlk = blockDim.x >> 5;
    const int waveGlobal = blockIdx.x * wavesPerBlk + waveInBlk;
    const int totalWaves = gridDim.x * wavesPerBlk;

    for (int i = tid; i < 8 * KTOPIC; i += blockDim.x) sCK[i] = 0.0f;
    __syncthreads();

    const int hh = lane >> 4;       // which half of the wave (K-slice selector)
    const int m  = lane & 15;       // A-matrix row owned by this lane
    const int n  = lane & 15;       // B/C-matrix column owned by this lane

    // Persistent f32 accumulators: C_j[t][n] counts words with time t, z=16j+n
    v8f c0 = {}, c1 = {}, c2 = {}, c3 = {};

    const int nBatches = (AW + 31) >> 5;
    for (int batch = waveGlobal; batch < nBatches; batch += totalWaves) {
        const int word  = (batch << 5) + lane;
        const bool valid = word < AW;

        int   t = valid ? tArr[word] : -1;
        int   d = valid ? dArr[word] : 0;
        int   w = valid ? wArr[word] : 0;
        int   z = valid ? zArr[word] : -1;
        float N = valid ? nArr[word] : 1.0f;

        // Low-contention scatter-adds straight to L2 (output fits in 192MB L2)
        if (valid) {
            const int dz = d * KTOPIC + z;
            atomicAdd(&outCDK[dz], 1.0f);
            atomicAdd(&outCWK[t * cwkPlaneStride + w * KTOPIC + z], 1.0f);
            atomicAdd(&outEta[dz], (1.0f + INIT_ALPHA) / (N + (float)KTOPIC * INIT_ALPHA));
        }

        // ---- CK via WMMA: C += onehot_t(16x32) x onehot_z(32x16) ----------
        // A layout (16-bit A 16x32, wave32): lanes 0-15 hold M=lane,
        // lanes 16-31 hold M=lane-16. VGPR g, slot s -> K = base(g)+8*hh+s,
        // base = {0,2,4,6,16,18,20,22}.
        v16h a;
#pragma unroll
        for (int i = 0; i < 16; ++i) {
            const int g = i >> 1, slot = i & 1;
            const int k = ((g < 4) ? (g * 2) : (16 + (g - 4) * 2)) + hh * 8 + slot;
            const int tk = __shfl(t, k, 32);
            a[i] = (tk == m) ? (_Float16)1.0f : (_Float16)0.0f;
        }
        // B layout (16-bit B 32x16, wave32): same K order as A;
        // element i -> K = 8*hh + i (i<8), 16 + 8*hh + (i-8) (i>=8), N = lane&15.
        int zk[16];
#pragma unroll
        for (int i = 0; i < 16; ++i) zk[i] = __shfl(z, hh * 8 + i + ((i >= 8) ? 8 : 0), 32);

        v16h b;
#pragma unroll
        for (int i = 0; i < 16; ++i) b[i] = (zk[i] == n) ? (_Float16)1.0f : (_Float16)0.0f;
        c0 = __builtin_amdgcn_wmma_f32_16x16x32_f16(false, a, false, b, (short)0, c0, false, false);
#pragma unroll
        for (int i = 0; i < 16; ++i) b[i] = (zk[i] == n + 16) ? (_Float16)1.0f : (_Float16)0.0f;
        c1 = __builtin_amdgcn_wmma_f32_16x16x32_f16(false, a, false, b, (short)0, c1, false, false);
#pragma unroll
        for (int i = 0; i < 16; ++i) b[i] = (zk[i] == n + 32) ? (_Float16)1.0f : (_Float16)0.0f;
        c2 = __builtin_amdgcn_wmma_f32_16x16x32_f16(false, a, false, b, (short)0, c2, false, false);
#pragma unroll
        for (int i = 0; i < 16; ++i) b[i] = (zk[i] == n + 48) ? (_Float16)1.0f : (_Float16)0.0f;
        c3 = __builtin_amdgcn_wmma_f32_16x16x32_f16(false, a, false, b, (short)0, c3, false, false);
    }

    // C layout (f32 16x16): lane covers N = lane&15; VGPR r holds M=r for
    // lanes 0-15 and M=r+8 for lanes 16-31. t < 8 => all valid counts live in
    // lanes 0-15, rows r=0..7. Reduce into the block's LDS CK tile.
    if (hh == 0) {
#pragma unroll
        for (int r = 0; r < 8; ++r) {
            atomicAdd(&sCK[r * KTOPIC + n],      c0[r]);
            atomicAdd(&sCK[r * KTOPIC + 16 + n], c1[r]);
            atomicAdd(&sCK[r * KTOPIC + 32 + n], c2[r]);
            atomicAdd(&sCK[r * KTOPIC + 48 + n], c3[r]);
        }
    }
    __syncthreads();

    // One batched global flush per block: 512 atomics instead of ~4M.
    for (int i = tid; i < 8 * KTOPIC; i += blockDim.x) {
        const float v = sCK[i];
        if (v != 0.0f) atomicAdd(&outCK[i], v);
    }
}

// ---------------------------------------------------------------------------
// Launcher
// ---------------------------------------------------------------------------
extern "C" void kernel_launch(void* const* d_in, const int* in_sizes, int n_in,
                              void* d_out, int out_size, void* d_ws, size_t ws_size,
                              hipStream_t stream) {
    (void)n_in; (void)d_ws; (void)ws_size;

    const int* tArr  = (const int*)  d_in[0];   // time_ind_per_word
    const int* dArr  = (const int*)  d_in[1];   // doc_indexes
    const int* wArr  = (const int*)  d_in[2];   // flatW
    const float* nAr = (const float*)d_in[3];   // N_per_word
    const int* zArr  = (const int*)  d_in[4];   // flatZ

    const int AW   = in_sizes[0];
    const int nCDK = in_sizes[5];               // AD*K
    const int nCWK = in_sizes[6];               // T*V*K
    const int nCK  = in_sizes[7];               // T*K
    const int T    = nCK / KTOPIC;              // = 8
    const int cwkPlaneStride = nCWK / T;        // = V*K

    float* out   = (float*)d_out;
    float* oCDK  = out;
    float* oCWK  = out + (size_t)nCDK;
    float* oCK   = out + (size_t)nCDK + (size_t)nCWK;
    float* oEta  = oCK + (size_t)nCK;

    // 1) zero the full output (poisoned by the harness)
    dtm_zero_kernel<<<2048, 256, 0, stream>>>(out, out_size);

    // 2) scatter-add with WMMA-based CK histogram
    dtm_scatter_kernel<<<1024, 256, 0, stream>>>(
        tArr, dArr, wArr, nAr, zArr,
        oCDK, oCWK, oCK, oEta, AW, cwkPlaneStride);
}
